// ModelLSTM_23819888623797
// MI455X (gfx1250) — compile-verified, matrix operand order checked by reading, not run on hardware
//
#include <hip/hip_runtime.h>
#include <hip/hip_bf16.h>

#define B_  32
#define T_  512
#define D_  512
#define H_  256
#define E_  64
#define R_  64
#define G4H 1024   // 4*H

#define AS1 __attribute__((address_space(1)))
#define AS3 __attribute__((address_space(3)))

typedef __attribute__((ext_vector_type(16))) _Float16 v16h;
typedef __attribute__((ext_vector_type(8)))  float    v8f;
typedef __attribute__((ext_vector_type(4)))  int      v4i;

union Frag16 { v16h v; float4 f4[2]; };

// A-frag (16x32, lane=M) and B-frag (32x16, lane=N) share this layout when the
// source is row-major [rows, ldk] f16: lane holds two contiguous 8-half runs.
__device__ __forceinline__ v16h ld_frag(const _Float16* __restrict__ base,
                                        int row, int ldk, int k0, int lane) {
  Frag16 fr;
  const _Float16* p = base + (size_t)row * ldk + k0 + ((lane >> 4) << 3);
  fr.f4[0] = *(const float4*)(p);
  fr.f4[1] = *(const float4*)(p + 16);
  return fr.v;
}

__device__ __forceinline__ v8f wmma16(v16h a, v16h b, v8f c) {
  return __builtin_amdgcn_wmma_f32_16x16x32_f16(false, a, false, b,
                                                (short)0, c, false, false);
}

__device__ __forceinline__ float sigm(float x) { return 1.f / (1.f + __expf(-x)); }

// ------- CDNA5 async copy global->LDS (ASYNCcnt path) -------
__device__ __forceinline__ void async_b128(const _Float16* g, _Float16* l) {
#if __has_builtin(__builtin_amdgcn_global_load_async_to_lds_b128)
  __builtin_amdgcn_global_load_async_to_lds_b128(
      (AS1 v4i*)const_cast<_Float16*>(g), (AS3 v4i*)l, 0, 0);
#else
  unsigned loff = (unsigned)(unsigned long long)l;
  asm volatile("global_load_async_to_lds_b128 %0, %1, off"
               :: "v"(loff), "v"(g) : "memory");
#endif
}
#if __has_builtin(__builtin_amdgcn_s_wait_asynccnt)
#define WAIT_ASYNC(n) __builtin_amdgcn_s_wait_asynccnt(n)
#else
#define WAIT_ASYNC(n) asm volatile("s_wait_asynccnt %0" :: "i"(n) : "memory")
#endif

// ---------------------------------------------------------------- utilities
__global__ void k_cvt(const float* __restrict__ s, _Float16* __restrict__ d, int n) {
  int i = blockIdx.x * 256 + threadIdx.x;
  if (i < n) d[i] = (_Float16)s[i];
}
__global__ void k_zero(float* __restrict__ d, int n) {
  int i = blockIdx.x * 256 + threadIdx.x;
  if (i < n) d[i] = 0.f;
}
__global__ void k_bias2(const float* __restrict__ a, const float* __restrict__ b,
                        float* __restrict__ o, int n) {
  int i = blockIdx.x * 256 + threadIdx.x;
  if (i < n) o[i] = a[i] + b[i];
}
// embedding gather -> f16 activations [B*T, D]
__global__ void k_embed(const int* __restrict__ sents, const float* __restrict__ emb,
                        _Float16* __restrict__ xh) {
  int bt = blockIdx.x;
  int tok = sents[bt];
  const float* e = emb + (size_t)tok * D_;
  _Float16* o = xh + (size_t)bt * D_;
  for (int d = threadIdx.x; d < D_; d += blockDim.x) o[d] = (_Float16)e[d];
}

// ------------------------------------------------- generic WMMA GEMM
// C[M,N] = A[M,K](f16) * W[N,K](f16)^T + bias ; optional relu / f16 copy.
// Block = 256 thr = 8 waves arranged 4(M)x2(N); block tile 128x64, k-step 32.
// A/B tiles are staged into LDS with async copies (double buffered): per
// k-step each thread issues 3 GLOBAL_LOAD_ASYNC_TO_LDS_B128; compute for step
// k overlaps the in-flight copies for step k+1 (s_wait_asynccnt 3).
__global__ __launch_bounds__(256) void k_gemm(
    const _Float16* __restrict__ A, const _Float16* __restrict__ W,
    const float* __restrict__ bias, float* __restrict__ C,
    _Float16* __restrict__ Ch, int M, int N, int K, int relu) {
  __shared__ _Float16 sA[2][128][40];
  __shared__ _Float16 sB[2][64][40];
  int tid = threadIdx.x, lane = tid & 31, w = tid >> 5;
  int wm = w >> 1, wn = w & 1;
  int nbn = N >> 6;
  int Mblk = (blockIdx.x / nbn) * 128;
  int Nblk = (blockIdx.x % nbn) * 64;

  auto stage = [&](int buf, int k0) {
#pragma unroll
    for (int q = 0; q < 2; q++) {               // A: 128 rows x 4 x 16B chunks
      int c = tid + q * 256;
      int r = c >> 2, k8 = (c & 3) << 3;
      async_b128(A + (size_t)(Mblk + r) * K + k0 + k8, &sA[buf][r][k8]);
    }
    int r = tid >> 2, k8 = (tid & 3) << 3;      // B: 64 rows x 4 x 16B chunks
    async_b128(W + (size_t)(Nblk + r) * K + k0 + k8, &sB[buf][r][k8]);
  };

  int nk = K >> 5;
  stage(0, 0);
  if (nk > 1) stage(1, 32);

  v8f acc[2][2] = {};
  for (int ks = 0; ks < nk; ks++) {
    int buf = ks & 1;
    if (ks + 1 < nk) { WAIT_ASYNC(3); } else { WAIT_ASYNC(0); }
    __syncthreads();                            // tiles for step ks visible
    v16h a0 = ld_frag(&sA[buf][0][0], wm * 32 + (lane & 15), 40, 0, lane);
    v16h a1 = ld_frag(&sA[buf][0][0], wm * 32 + 16 + (lane & 15), 40, 0, lane);
    v16h b0 = ld_frag(&sB[buf][0][0], wn * 32 + (lane & 15), 40, 0, lane);
    v16h b1 = ld_frag(&sB[buf][0][0], wn * 32 + 16 + (lane & 15), 40, 0, lane);
    acc[0][0] = wmma16(a0, b0, acc[0][0]);
    acc[0][1] = wmma16(a0, b1, acc[0][1]);
    acc[1][0] = wmma16(a1, b0, acc[1][0]);
    acc[1][1] = wmma16(a1, b1, acc[1][1]);
    __syncthreads();                            // all waves done with buf
    if (ks + 2 < nk) stage(buf, (ks + 2) * 32);
  }

  int half8 = (lane >> 4) * 8, ncol = lane & 15;
#pragma unroll
  for (int i = 0; i < 2; i++)
#pragma unroll
    for (int j = 0; j < 2; j++) {
      int n = Nblk + wn * 32 + j * 16 + ncol;
      float bv = bias ? bias[n] : 0.f;
#pragma unroll
      for (int v = 0; v < 8; v++) {
        int m = Mblk + wm * 32 + i * 16 + half8 + v;
        float val = acc[i][j][v] + bv;
        if (relu) val = fmaxf(val, 0.f);
        C[(size_t)m * N + n] = val;
        if (Ch) Ch[(size_t)m * N + n] = (_Float16)val;
      }
    }
}

// ------------------------------------------------- LSTM recurrence (per dir)
// grid=2 (dir), 1024 thr = 32 waves. Wave w: mtile=w&1 (batch 16-rows),
// jg=w>>1 (16 h-columns). Whh B-fragments live in VGPRs across the t-loop;
// h ping-pongs in LDS as f16. Z holds precomputed x*Wih^T + (bih+bhh).
__global__ __launch_bounds__(1024) void k_lstm(
    const float* __restrict__ Zf, const float* __restrict__ Zb,
    const _Float16* __restrict__ WhhF, const _Float16* __restrict__ WhhB,
    float* __restrict__ Y, _Float16* __restrict__ Yh) {
  __shared__ _Float16 hbuf[2][32][264];
  int dir = blockIdx.x;
  const float* Z = dir ? Zb : Zf;
  const _Float16* Whh = dir ? WhhB : WhhF;
  int tid = threadIdx.x, lane = tid & 31, w = tid >> 5;
  int mtile = w & 1, jg = w >> 1;
  int half8 = (lane >> 4) * 8, ncol = lane & 15;

  for (int i = tid; i < 32 * 264; i += 1024)
    hbuf[0][i / 264][i % 264] = (_Float16)0.f;
  __syncthreads();

  v16h bfr[4][8];
#pragma unroll
  for (int g = 0; g < 4; g++)
#pragma unroll
    for (int kc = 0; kc < 8; kc++)
      bfr[g][kc] = ld_frag(Whh, g * 256 + jg * 16 + (lane & 15), H_, kc * 32, lane);

  float creg[8] = {0.f, 0.f, 0.f, 0.f, 0.f, 0.f, 0.f, 0.f};

  for (int t = 0; t < T_; t++) {
    int tt = dir ? (T_ - 1 - t) : t;
    int cur = t & 1, nxt = cur ^ 1;
    v16h afr[8];
#pragma unroll
    for (int kc = 0; kc < 8; kc++)
      afr[kc] = ld_frag(&hbuf[cur][0][0], mtile * 16 + (lane & 15), 264, kc * 32, lane);
    v8f acc[4];
#pragma unroll
    for (int g = 0; g < 4; g++)
#pragma unroll
      for (int v = 0; v < 8; v++) {
        int b = mtile * 16 + half8 + v;
        acc[g][v] = Z[((size_t)b * T_ + tt) * G4H + g * 256 + jg * 16 + ncol];
      }
#pragma unroll
    for (int kc = 0; kc < 8; kc++)
#pragma unroll
      for (int g = 0; g < 4; g++)
        acc[g] = wmma16(afr[kc], bfr[g][kc], acc[g]);
#pragma unroll
    for (int v = 0; v < 8; v++) {
      float ig = sigm(acc[0][v]);
      float fg = sigm(acc[1][v]);
      float gg = tanhf(acc[2][v]);
      float og = sigm(acc[3][v]);
      float cn = fg * creg[v] + ig * gg;
      creg[v] = cn;
      float hn = og * tanhf(cn);
      int b = mtile * 16 + half8 + v;
      int col = jg * 16 + ncol;
      hbuf[nxt][b][col] = (_Float16)hn;
      size_t yo = ((size_t)b * T_ + tt) * D_ + dir * H_ + col;
      Y[yo] = hn;
      Yh[yo] = (_Float16)hn;
    }
    __syncthreads();
  }
}

// ------------------------------------------------- cumsum over T
__global__ void k_cumsum(const float* __restrict__ Y, float* __restrict__ cs) {
  int idx = blockIdx.x * 256 + threadIdx.x;   // b*D + d
  int b = idx >> 9, d = idx & 511;
  float* c = cs + (size_t)b * (T_ + 1) * D_;
  const float* y = Y + (size_t)b * T_ * D_;
  float a = 0.f;
  c[d] = 0.f;
  for (int t = 0; t < T_; t++) {
    a += y[(size_t)t * D_ + d];
    c[(size_t)(t + 1) * D_ + d] = a;
  }
}

// ------------------------------------------------- ragged span pool (scatter)
__global__ __launch_bounds__(256) void k_pool(const int* __restrict__ ent,
                                              const float* __restrict__ cs,
                                              float* __restrict__ cont,
                                              float* __restrict__ mask) {
  int b = blockIdx.x / E_, e = blockIdx.x % E_;
  const int* ei = ent + (b * E_ + e) * 3;
  int id = ei[0], st = ei[1], en = ei[2];
  if (id < 0) return;
  float inv = 1.f / fmaxf((float)(en - st), 1.f);
  const float* c = cs + (size_t)b * (T_ + 1) * D_;
  for (int d = threadIdx.x; d < D_; d += 256) {
    float s = c[(size_t)en * D_ + d] - c[(size_t)st * D_ + d];
    atomicAdd(&cont[((size_t)b * E_ + id) * D_ + d], s * inv);
  }
  if (threadIdx.x == 0) atomicAdd(&mask[b * E_ + id], 1.f);
}

// ------------------------------------------------- LayerNorm -> f16
__global__ __launch_bounds__(256) void k_ln(const float* __restrict__ cont,
                                            const float* __restrict__ g,
                                            const float* __restrict__ bb,
                                            _Float16* __restrict__ cwh) {
  __shared__ float red[256];
  int row = blockIdx.x, tid = threadIdx.x;
  const float* x = cont + (size_t)row * D_;
  float x0 = x[tid], x1 = x[tid + 256];
  red[tid] = x0 + x1;
  __syncthreads();
  for (int o = 128; o > 0; o >>= 1) { if (tid < o) red[tid] += red[tid + o]; __syncthreads(); }
  float mu = red[0] / 512.f;
  __syncthreads();
  float d0 = x0 - mu, d1 = x1 - mu;
  red[tid] = d0 * d0 + d1 * d1;
  __syncthreads();
  for (int o = 128; o > 0; o >>= 1) { if (tid < o) red[tid] += red[tid + o]; __syncthreads(); }
  float inv = rsqrtf(red[0] / 512.f + 1e-5f);
  cwh[(size_t)row * D_ + tid]       = (_Float16)(d0 * inv * g[tid] + bb[tid]);
  cwh[(size_t)row * D_ + tid + 256] = (_Float16)(d1 * inv * g[tid + 256] + bb[tid + 256]);
}

// ------------------------------------------------- fused pairwise block
// Block (b,i): rows j=0..63. A = relu(rel1[b,j]+rel2[b,i]) -> LDS f16,
// H2 = relu(A @ projW^T + pb) (LDS reused), logits = H2 @ decW^T + db,
// mask, log-softmax over R. 512 thr = 16 waves.
__global__ __launch_bounds__(512) void k_pair(
    const float* __restrict__ rel1, const float* __restrict__ rel2,
    const _Float16* __restrict__ projW, const float* __restrict__ pb,
    const _Float16* __restrict__ decW, const float* __restrict__ db,
    const float* __restrict__ mask, float* __restrict__ out) {
  __shared__ _Float16 Ah[64][520];
  __shared__ float Lg[64][68];
  int b = blockIdx.x / E_, i = blockIdx.x % E_;
  int tid = threadIdx.x, lane = tid & 31, w = tid >> 5;
  int half8 = (lane >> 4) * 8, ncol = lane & 15;

  const float* r2 = rel2 + ((size_t)(b * E_ + i)) * D_;
  for (int idx = tid; idx < 64 * D_; idx += 512) {
    int j = idx >> 9, d = idx & 511;
    float v = rel1[((size_t)(b * E_ + j)) * D_ + d] + r2[d];
    Ah[j][d] = (_Float16)fmaxf(v, 0.f);
  }
  __syncthreads();

  // stage 1: 64x512, wave w -> Mtile (w&3), N range (w>>2)*128..+127
  int wm = w & 3, wn = w >> 2;
  v8f acc[8] = {};
  for (int kc = 0; kc < 16; kc++) {
    v16h a = ld_frag(&Ah[0][0], wm * 16 + (lane & 15), 520, kc * 32, lane);
    if (kc + 1 < 16)   // hint the next weight slice toward the caches
      __builtin_prefetch(projW + (size_t)(wn * 128 + (lane & 15)) * D_ + (kc + 1) * 32, 0, 1);
#pragma unroll
    for (int bi = 0; bi < 8; bi++) {
      v16h bb = ld_frag(projW, wn * 128 + bi * 16 + (lane & 15), D_, kc * 32, lane);
      acc[bi] = wmma16(a, bb, acc[bi]);
    }
  }
  __syncthreads();   // all waves done reading Ah
#pragma unroll
  for (int bi = 0; bi < 8; bi++) {
    int n = wn * 128 + bi * 16 + ncol;
    float bv = pb[n];
#pragma unroll
    for (int v = 0; v < 8; v++) {
      int j = wm * 16 + half8 + v;
      Ah[j][n] = (_Float16)fmaxf(acc[bi][v] + bv, 0.f);   // reuse LDS as H2
    }
  }
  __syncthreads();

  // stage 2: 64x64 logits, wave w -> (jm=w&3, rn=w>>2)
  v8f aL = {};
  for (int kc = 0; kc < 16; kc++) {
    v16h a = ld_frag(&Ah[0][0], (w & 3) * 16 + (lane & 15), 520, kc * 32, lane);
    v16h bb = ld_frag(decW, (w >> 2) * 16 + (lane & 15), D_, kc * 32, lane);
    aL = wmma16(a, bb, aL);
  }
  float mi = fminf(mask[b * E_ + i], 1.f);
#pragma unroll
  for (int v = 0; v < 8; v++) {
    int j = (w & 3) * 16 + half8 + v;
    int r = (w >> 2) * 16 + ncol;
    float mj = fminf(mask[b * E_ + j], 1.f);
    Lg[j][r] = (aL[v] + db[r]) * mi * mj;
  }
  __syncthreads();

  if (tid < 64) {
    int j = tid;
    float mx = -3.4e38f;
    for (int r = 0; r < R_; r++) mx = fmaxf(mx, Lg[j][r]);
    float se = 0.f;
    for (int r = 0; r < R_; r++) se += __expf(Lg[j][r] - mx);
    float lse = mx + __logf(se);
    float* op = out + (((size_t)(b * E_ + i)) * E_ + j) * R_;
    for (int r = 0; r < R_; r++) op[r] = Lg[j][r] - lse;
  }
}

// =================================================================== launch
extern "C" void kernel_launch(void* const* d_in, const int* in_sizes, int n_in,
                              void* d_out, int out_size, void* d_ws, size_t ws_size,
                              hipStream_t stream) {
  const int*   sents  = (const int*)d_in[0];
  const int*   ent    = (const int*)d_in[1];
  const float* emb    = (const float*)d_in[3];
  const float* Wih    = (const float*)d_in[4];
  const float* Whh    = (const float*)d_in[5];
  const float* bih    = (const float*)d_in[6];
  const float* bhh    = (const float*)d_in[7];
  const float* ln_g   = (const float*)d_in[8];
  const float* ln_b   = (const float*)d_in[9];
  const float* rel1_w = (const float*)d_in[10];
  const float* rel1_b = (const float*)d_in[11];
  const float* rel2_w = (const float*)d_in[12];
  const float* rel2_b = (const float*)d_in[13];
  const float* proj_w = (const float*)d_in[14];
  const float* proj_b = (const float*)d_in[15];
  const float* dec_w  = (const float*)d_in[16];
  const float* dec_b  = (const float*)d_in[17];
  float* out = (float*)d_out;

  char* p = (char*)d_ws;
  auto alloc = [&](size_t bytes) -> void* {
    char* r = p; p += (bytes + 255) & ~(size_t)255; return (void*)r;
  };
  const size_t NBT = (size_t)B_ * T_;            // 16384
  _Float16* WihH  = (_Float16*)alloc(2ull * 2 * G4H * D_ * 2);
  _Float16* WhhH  = (_Float16*)alloc(2ull * 2 * G4H * H_ * 2);
  _Float16* r1wH  = (_Float16*)alloc((size_t)D_ * D_ * 2);
  _Float16* r2wH  = (_Float16*)alloc((size_t)D_ * D_ * 2);
  _Float16* prwH  = (_Float16*)alloc((size_t)D_ * D_ * 2);
  _Float16* dcwH  = (_Float16*)alloc((size_t)R_ * D_ * 2);
  float*    bsum  = (float*)alloc(2ull * 2 * G4H * 4);
  _Float16* x0h   = (_Float16*)alloc(NBT * D_ * 2);
  float*    Zf    = (float*)alloc(NBT * G4H * 4);
  float*    Zb    = (float*)alloc(NBT * G4H * 4);
  float*    y1    = (float*)alloc(NBT * D_ * 4);
  _Float16* y1h   = (_Float16*)alloc(NBT * D_ * 2);
  float*    y2    = (float*)alloc(NBT * D_ * 4);
  _Float16* y2h   = (_Float16*)alloc(NBT * D_ * 2);
  float*    csum  = (float*)alloc((size_t)B_ * (T_ + 1) * D_ * 4);
  float*    cont  = (float*)alloc((size_t)B_ * E_ * D_ * 4);
  float*    mask  = (float*)alloc((size_t)B_ * E_ * 4);
  _Float16* cwh   = (_Float16*)alloc((size_t)B_ * E_ * D_ * 2);
  float*    rel1  = (float*)alloc((size_t)B_ * E_ * D_ * 4);
  float*    rel2  = (float*)alloc((size_t)B_ * E_ * D_ * 4);

  auto cvt = [&](const float* s, _Float16* d, int n) {
    k_cvt<<<(n + 255) / 256, 256, 0, stream>>>(s, d, n);
  };
  cvt(Wih, WihH, 2 * 2 * G4H * D_);
  cvt(Whh, WhhH, 2 * 2 * G4H * H_);
  cvt(rel1_w, r1wH, D_ * D_);
  cvt(rel2_w, r2wH, D_ * D_);
  cvt(proj_w, prwH, D_ * D_);
  cvt(dec_w, dcwH, R_ * D_);
  k_bias2<<<(2 * 2 * G4H + 255) / 256, 256, 0, stream>>>(bih, bhh, bsum, 2 * 2 * G4H);
  k_embed<<<(int)NBT, 256, 0, stream>>>(sents, emb, x0h);
  k_zero<<<((B_ * E_ * D_) + 255) / 256, 256, 0, stream>>>(cont, B_ * E_ * D_);
  k_zero<<<(B_ * E_ + 255) / 256, 256, 0, stream>>>(mask, B_ * E_);

  auto gemm = [&](const _Float16* A, const _Float16* W, const float* bias,
                  float* C, _Float16* Ch, int M, int N, int K, int relu) {
    int grid = (M / 128) * (N / 64);
    k_gemm<<<grid, 256, 0, stream>>>(A, W, bias, C, Ch, M, N, K, relu);
  };

  // layer 0: input projections + recurrence
  gemm(x0h, WihH + 0ull * G4H * D_, bsum + 0 * G4H, Zf, nullptr, (int)NBT, G4H, D_, 0);
  gemm(x0h, WihH + 1ull * G4H * D_, bsum + 1 * G4H, Zb, nullptr, (int)NBT, G4H, D_, 0);
  k_lstm<<<2, 1024, 0, stream>>>(Zf, Zb, WhhH + 0ull * G4H * H_, WhhH + 1ull * G4H * H_, y1, y1h);
  // layer 1
  gemm(y1h, WihH + 2ull * G4H * D_, bsum + 2 * G4H, Zf, nullptr, (int)NBT, G4H, D_, 0);
  gemm(y1h, WihH + 3ull * G4H * D_, bsum + 3 * G4H, Zb, nullptr, (int)NBT, G4H, D_, 0);
  k_lstm<<<2, 1024, 0, stream>>>(Zf, Zb, WhhH + 2ull * G4H * H_, WhhH + 3ull * G4H * H_, y2, y2h);

  k_cumsum<<<(B_ * D_) / 256, 256, 0, stream>>>(y2, csum);
  k_pool<<<B_ * E_, 256, 0, stream>>>(ent, csum, cont, mask);
  k_ln<<<B_ * E_, 256, 0, stream>>>(cont, ln_g, ln_b, cwh);

  gemm(cwh, r1wH, rel1_b, rel1, nullptr, B_ * E_, D_, D_, 0);
  gemm(cwh, r2wH, rel2_b, rel2, nullptr, B_ * E_, D_, D_, 0);

  k_pair<<<B_ * E_, 512, 0, stream>>>(rel1, rel2, prwH, proj_b, dcwH, dec_b, mask, out);
}
